// Descriptor_Net_23347442221745
// MI455X (gfx1250) — compile-verified, matrix operand order checked by reading, not run on hardware
//
#include <hip/hip_runtime.h>
#include <hip/hip_bf16.h>

// ---------------- problem constants ----------------
#define RAD_N 9
#define AZI_N 60
#define ELE_N 30
#define S_VOX (RAD_N * ELE_N * AZI_N)   // 16200
#define NSAMP 30
#define NPTS 2048
#define BATCH 4
#define HW 1800                          // ELE_N * AZI_N
#define PH 32                            // padded ele rows (1 zero row each side)
#define PW 62                            // padded azi cols (wrap col each side)
#define PHW (PH * PW)                    // 1984
#define EPSV 1e-5f
#define PI_F 3.14159265358979f
#define TWO_PI_F 6.28318530717959f

// ---------------- WMMA fragment types ----------------
typedef __attribute__((ext_vector_type(16))) _Float16 v16h;
typedef __attribute__((ext_vector_type(8)))  float    v8f;

union AFrag { v16h v; _Float16 h[16]; };
union CFrag { v8f  v; float    f[8]; };

// ---------------- workspace layout (bytes) ----------------
static const size_t OFF_R    = 0;          // BATCH*9 f32
static const size_t OFF_PSUM = 256;        // 9 f32
static const size_t OFF_RSS  = 512;        // 32 f32
static const size_t OFF_MEAN = 768;        // 128 f32
static const size_t OFF_ISTD = 1280;       // 128 f32
static const size_t OFF_DROT = 2048;       // BATCH*NPTS*3 f32 (96 KB)
static const size_t OFF_NEWP = 102400;     // BATCH*S_VOX*NSAMP*3 f32 (23.3 MB)
static const size_t OFF_ACT0 = 23430400;   // padded f16, 2,539,520 halfs max (5.08 MB)
static const size_t OFF_ACT1 = 28509440;
static const size_t OFF_CONV = 33588480;   // f32 conv out, 2,304,000 f32 (9.2 MB)
static const size_t OFF_WPK  = 42804480;   // packed weights, 221,184 halfs max

// ---------------- tiny utility kernels ----------------
__global__ void zero_kernel(float* p, int n) {
    int i = blockIdx.x * blockDim.x + threadIdx.x;
    if (i < n) p[i] = 0.f;
}

// zero the two ele-pad rows of a padded activation buffer: (BATCH*CD) slices
__global__ void padzero_kernel(_Float16* act, int CD) {
    int i = blockIdx.x * blockDim.x + threadIdx.x;
    int total = BATCH * CD * 2 * PW;
    if (i >= total) return;
    int col  = i % PW;
    int rsel = (i / PW) & 1;
    int bcd  = i / (2 * PW);
    act[(size_t)bcd * PHW + (rsel ? (PH - 1) * PW : 0) + col] = (_Float16)0.f;
}

// repack weights (co, ci, tap) f32 -> WMMA B-fragment order, k = tap*Ci + ci
// wpack[((cotile*nchunk + ch)*32 + lane)*16 + j]
__global__ void wpack_kernel(const float* __restrict__ w, _Float16* __restrict__ wpack,
                             int Ci, int ciLog2, int K9, int nchunk, int total) {
    int t = blockIdx.x * blockDim.x + threadIdx.x;
    if (t >= total) return;
    int j      = t & 15;
    int lane   = (t >> 4) & 31;
    int ch     = (t >> 9) % nchunk;
    int cotile = (t >> 9) / nchunk;
    int klocal = ((j >> 3) << 4) + (((lane >> 4) & 1) << 3) + (j & 7);
    int k  = ch * 32 + klocal;
    int co = cotile * 16 + (lane & 15);
    _Float16 val = (_Float16)0.f;
    if (k < Ci * K9) {
        int ci  = k & (Ci - 1);
        int tap = k >> ciLog2;
        val = (_Float16)w[((size_t)co * Ci + ci) * K9 + tap];
    }
    wpack[t] = val;
}

// ---------------- covariance + eigen + rodrigues ----------------
__global__ void cov_eig_kernel(const float* __restrict__ x, float* __restrict__ Rm) {
    int b = blockIdx.x;
    __shared__ float red[6 * 256];
    const float* xb = x + (size_t)b * NPTS * 3;
    float cx = xb[(NPTS - 1) * 3 + 0];
    float cy = xb[(NPTS - 1) * 3 + 1];
    float cz = xb[(NPTS - 1) * 3 + 2];
    float s[6] = {0.f, 0.f, 0.f, 0.f, 0.f, 0.f};
    for (int n = threadIdx.x; n < NPTS; n += blockDim.x) {
        float dx = xb[n * 3 + 0] - cx;
        float dy = xb[n * 3 + 1] - cy;
        float dz = xb[n * 3 + 2] - cz;
        s[0] += dx * dx; s[1] += dy * dy; s[2] += dz * dz;
        s[3] += dx * dy; s[4] += dx * dz; s[5] += dy * dz;
    }
    for (int k = 0; k < 6; k++) red[k * 256 + threadIdx.x] = s[k];
    __syncthreads();
    for (int off = 128; off > 0; off >>= 1) {
        if ((int)threadIdx.x < off)
            for (int k = 0; k < 6; k++)
                red[k * 256 + threadIdx.x] += red[k * 256 + threadIdx.x + off];
        __syncthreads();
    }
    if (threadIdx.x == 0) {
        float a00 = red[0], a11 = red[256], a22 = red[512];
        float a01 = red[768], a02 = red[1024], a12 = red[1280];
        float p1 = a01 * a01 + a02 * a02 + a12 * a12;
        float vx, vy, vz;
        if (p1 < 1e-10f) {
            vx = (a00 <= a11 && a00 <= a22) ? 1.f : 0.f;
            vy = (vx == 0.f && a11 <= a22) ? 1.f : 0.f;
            vz = (vx == 0.f && vy == 0.f) ? 1.f : 0.f;
        } else {
            float q = (a00 + a11 + a22) / 3.f;
            float p2 = (a00 - q) * (a00 - q) + (a11 - q) * (a11 - q) +
                       (a22 - q) * (a22 - q) + 2.f * p1;
            float pp = sqrtf(p2 / 6.f);
            float b00 = (a00 - q) / pp, b11 = (a11 - q) / pp, b22 = (a22 - q) / pp;
            float b01 = a01 / pp, b02 = a02 / pp, b12 = a12 / pp;
            float detB = b00 * (b11 * b22 - b12 * b12)
                       - b01 * (b01 * b22 - b12 * b02)
                       + b02 * (b01 * b12 - b11 * b02);
            float r = fminf(1.f, fmaxf(-1.f, 0.5f * detB));
            float phi = acosf(r) / 3.f;
            float e3 = q + 2.f * pp * cosf(phi + 2.09439510239f); // smallest eigenvalue
            float r0x = a00 - e3, r0y = a01, r0z = a02;
            float r1x = a01, r1y = a11 - e3, r1z = a12;
            float r2x = a02, r2y = a12, r2z = a22 - e3;
            float c0x = r0y * r1z - r0z * r1y, c0y = r0z * r1x - r0x * r1z, c0z = r0x * r1y - r0y * r1x;
            float c1x = r1y * r2z - r1z * r2y, c1y = r1z * r2x - r1x * r2z, c1z = r1x * r2y - r1y * r2x;
            float c2x = r2y * r0z - r2z * r0y, c2y = r2z * r0x - r2x * r0z, c2z = r2x * r0y - r2y * r0x;
            float n0 = c0x * c0x + c0y * c0y + c0z * c0z;
            float n1 = c1x * c1x + c1y * c1y + c1z * c1z;
            float n2 = c2x * c2x + c2y * c2y + c2z * c2z;
            if (n0 >= n1 && n0 >= n2)      { vx = c0x; vy = c0y; vz = c0z; }
            else if (n1 >= n2)             { vx = c1x; vy = c1y; vz = c1z; }
            else                           { vx = c2x; vy = c2y; vz = c2z; }
        }
        float nn = sqrtf(vx * vx + vy * vy + vz * vz) + 1e-8f;
        vx /= nn; vy /= nn; vz /= nn;
        if ((-vx * cx - vy * cy - vz * cz) < 0.f) { vx = -vx; vy = -vy; vz = -vz; }
        nn = sqrtf(vx * vx + vy * vy + vz * vz) + 1e-8f;
        vx /= nn; vy /= nn; vz /= nn;
        float ax = vy, ay = -vx;                     // v x e3
        float an = sqrtf(ax * ax + ay * ay) + 1e-8f;
        ax /= an; ay /= an;
        float c = fminf(1.f - 1e-6f, fmaxf(-1.f + 1e-6f, vz));
        float th = acosf(c);
        float st = sinf(th), omc = 1.f - cosf(th);
        float* M = Rm + b * 9;
        M[0] = 1.f + omc * (-ay * ay);
        M[1] = omc * (ax * ay);
        M[2] = st * ay;
        M[3] = omc * (ax * ay);
        M[4] = 1.f + omc * (-ax * ax);
        M[5] = st * (-ax);
        M[6] = st * (-ay);
        M[7] = st * (ax);
        M[8] = 1.f + omc * (-(ax * ax + ay * ay));
    }
}

// ---------------- rotate deltas: p' = M p ----------------
__global__ void rotate_kernel(const float* __restrict__ x, const float* __restrict__ Rm,
                              float* __restrict__ drot) {
    int i = blockIdx.x * blockDim.x + threadIdx.x;
    if (i >= BATCH * NPTS) return;
    int b = i / NPTS, n = i % NPTS;
    const float* xb = x + (size_t)b * NPTS * 3;
    float cx = xb[(NPTS - 1) * 3 + 0];
    float cy = xb[(NPTS - 1) * 3 + 1];
    float cz = xb[(NPTS - 1) * 3 + 2];
    float dx = xb[n * 3 + 0] - cx;
    float dy = xb[n * 3 + 1] - cy;
    float dz = xb[n * 3 + 2] - cz;
    const float* M = Rm + b * 9;
    drot[(size_t)i * 3 + 0] = M[0] * dx + M[1] * dy + M[2] * dz;
    drot[(size_t)i * 3 + 1] = M[3] * dx + M[4] * dy + M[5] * dz;
    drot[(size_t)i * 3 + 2] = M[6] * dx + M[7] * dy + M[8] * dz;
}

// ---------------- grouping: one wave per (b, voxel) ----------------
__global__ void group_kernel(const float* __restrict__ drot, float* __restrict__ newp,
                             float* __restrict__ psum) {
    int row = blockIdx.x;                 // b*S_VOX + s
    int lane = threadIdx.x;               // 0..31
    int b = row / S_VOX, s = row % S_VOX;
    int ir = s / (ELE_N * AZI_N);
    int rem = s % (ELE_N * AZI_N);
    int ie = rem / AZI_N, ia = rem % AZI_N;
    float rr = 2.0f * ((float)ir + 0.5f) / (float)RAD_N;
    float el = PI_F * ((float)ie + 0.5f) / (float)ELE_N;
    float az = TWO_PI_F * ((float)ia + 0.5f) / (float)AZI_N;
    float se = sinf(el);
    float vxg = rr * se * cosf(az);
    float vyg = rr * se * sinf(az);
    float vzg = rr * cosf(el);
    float rotang = TWO_PI_F * (float)ia / (float)AZI_N;
    float ca = cosf(rotang), sa = sinf(rotang);

    const float* db = drot + (size_t)b * NPTS * 3;
    float* op = newp + (size_t)row * NSAMP * 3;

    int cnt = 0;
    float sx = 0, sy = 0, sz = 0;
    float sxx = 0, syy = 0, szz = 0, sxy = 0, sxz = 0, syz = 0;

    for (int base = 0; base < NPTS && cnt < NSAMP; base += 32) {
        int n = base + lane;
        float px = db[(size_t)n * 3 + 0];
        float py = db[(size_t)n * 3 + 1];
        float pz = db[(size_t)n * 3 + 2];
        float ddx = px - vxg, ddy = py - vyg, ddz = pz - vzg;
        float d2 = ddx * ddx + ddy * ddy + ddz * ddz;
        unsigned long long m = __ballot(d2 <= 0.09f);
        while (m != 0ull && cnt < NSAMP) {
            int l = __builtin_ctzll(m);
            m &= (m - 1ull);
            float qx = __shfl(ddx, l);
            float qy = __shfl(ddy, l);
            float qz = __shfl(ddz, l);
            float nx = qx * ca + qy * sa;
            float ny = -qx * sa + qy * ca;
            float nz = qz;
            if (lane == 0) {
                op[cnt * 3 + 0] = nx;
                op[cnt * 3 + 1] = ny;
                op[cnt * 3 + 2] = nz;
            }
            sx += nx; sy += ny; sz += nz;
            sxx += nx * nx; syy += ny * ny; szz += nz * nz;
            sxy += nx * ny; sxz += nx * nz; syz += ny * nz;
            cnt++;
        }
    }
    if (lane == 0) {
        for (int k = cnt; k < NSAMP; k++) {
            op[k * 3 + 0] = 0.f; op[k * 3 + 1] = 0.f; op[k * 3 + 2] = 0.f;
        }
        atomicAdd(&psum[0], sx);  atomicAdd(&psum[1], sy);  atomicAdd(&psum[2], sz);
        atomicAdd(&psum[3], sxx); atomicAdd(&psum[4], syy); atomicAdd(&psum[5], szz);
        atomicAdd(&psum[6], sxy); atomicAdd(&psum[7], sxz); atomicAdd(&psum[8], syz);
    }
}

// ---------------- analytic BN stats for the raise MLP ----------------
__global__ void raise_stats_kernel(const float* __restrict__ psum, const float* __restrict__ wr,
                                   const float* __restrict__ br, const float* __restrict__ g,
                                   const float* __restrict__ be, float* __restrict__ rss) {
    int o = threadIdx.x;
    if (o >= 16) return;
    float cnt = (float)(BATCH * S_VOX * NSAMP);
    float w0 = wr[0 * 16 + o], w1 = wr[1 * 16 + o], w2 = wr[2 * 16 + o];
    float sp0 = psum[0], sp1 = psum[1], sp2 = psum[2];
    float sxx = psum[3], syy = psum[4], szz = psum[5];
    float sxy = psum[6], sxz = psum[7], syz = psum[8];
    float bo = br[o];
    float lin = (w0 * sp0 + w1 * sp1 + w2 * sp2) / cnt;
    float mean = lin + bo;
    float e2 = (w0 * w0 * sxx + w1 * w1 * syy + w2 * w2 * szz +
                2.f * (w0 * w1 * sxy + w0 * w2 * sxz + w1 * w2 * syz)) / cnt +
               2.f * bo * lin + bo * bo;
    float var = e2 - mean * mean;
    if (var < 0.f) var = 0.f;
    float scale = g[o] * rsqrtf(var + EPSV);
    rss[o] = scale;
    rss[16 + o] = be[o] - mean * scale;
}

// ---------------- raise MLP + BN + relu + max over samples (padded write) ----------------
__global__ void raise_max_kernel(const float* __restrict__ newp, const float* __restrict__ wr,
                                 const float* __restrict__ br, const float* __restrict__ rss,
                                 _Float16* __restrict__ act) {
    int i = blockIdx.x * blockDim.x + threadIdx.x;
    if (i >= BATCH * S_VOX * 16) return;
    int o = i & 15;
    int t = i >> 4;
    int s = t % S_VOX;
    int b = t / S_VOX;
    float w0 = wr[o], w1 = wr[16 + o], w2 = wr[32 + o];
    float bo = br[o];
    float scale = rss[o], shift = rss[16 + o];
    const float* p = newp + ((size_t)(b * S_VOX + s)) * NSAMP * 3;
    float mx = 0.f;
    for (int n = 0; n < NSAMP; n++) {
        float h = p[n * 3 + 0] * w0 + p[n * 3 + 1] * w1 + p[n * 3 + 2] * w2 + bo;
        float v = fmaxf(h * scale + shift, 0.f);
        mx = fmaxf(mx, v);
    }
    int ir = s / HW;
    int rest = s % HW;
    int ie = rest / AZI_N, ia = rest % AZI_N;
    size_t base = (((size_t)b * 16 + o) * RAD_N + ir) * PHW;
    _Float16 hv = (_Float16)mx;
    act[base + (size_t)(ie + 1) * PW + (ia + 1)] = hv;
    if (ia == 0)           act[base + (size_t)(ie + 1) * PW + (PW - 1)] = hv;  // wrap right
    if (ia == AZI_N - 1)   act[base + (size_t)(ie + 1) * PW + 0] = hv;         // wrap left
}

// ---------------- implicit-GEMM conv via WMMA f16, 2x2 register tiling ----------------
// Fully-specialized per layer: all dims compile-time -> no runtime div/mod chains.
// in   : padded f16 (BATCH, CI, DIN, 32, 62)
// wpack: B-fragment packed f16 [(CO/16)][NCHUNK][lane][16], k = tap*CI + ci
// out  : f32 (BATCH, CO, DOUT, 1800) pre-BN (bias included)
template <int CI, int CILOG2, int CO, int DIN, int DOUT, int KD>
__global__ void conv_wmma_kernel(const _Float16* __restrict__ in, const v16h* __restrict__ wpack,
                                 const float* __restrict__ bias, float* __restrict__ out) {
    constexpr int K9 = KD * 9;
    constexpr int K = CI * K9;
    constexpr int NCHUNK = (K + 31) / 32;
    constexpr int DINPHW = DIN * PHW;

    int lane  = threadIdx.x;
    int tilex = blockIdx.x;               // 57 tiles of 32 spatial positions
    int cot   = blockIdx.y;               // CO/32 pairs of co-tiles
    int bz    = blockIdx.z;               // b*DOUT + d
    int b = bz / DOUT, d = bz % DOUT;     // compile-time divisor
    int mrow  = lane & 15;
    int khalf = (lane >> 4) << 3;

    // two spatial tiles
    int pos0 = tilex * 32 + mrow;
    int pos1 = pos0 + 16;
    bool v0 = pos0 < HW, v1 = pos1 < HW;
    int ho0 = v0 ? pos0 / AZI_N : 0, wo0 = v0 ? pos0 % AZI_N : 0;
    int ho1 = v1 ? pos1 / AZI_N : 0, wo1 = v1 ? pos1 % AZI_N : 0;
    size_t base_bd = ((size_t)b * CI * DIN + d) * PHW;
    size_t lanebase0 = base_bd + (size_t)ho0 * PW + wo0;
    size_t lanebase1 = base_bd + (size_t)ho1 * PW + wo1;

    // two co tiles
    int co0 = cot * 32 + mrow;
    int co1 = co0 + 16;
    const v16h* wp0 = wpack + ((size_t)(cot * 2 + 0) * NCHUNK) * 32 + lane;
    const v16h* wp1 = wpack + ((size_t)(cot * 2 + 1) * NCHUNK) * 32 + lane;

    CFrag acc00, acc01, acc10, acc11;
#pragma unroll
    for (int v = 0; v < 8; v++) {
        acc00.f[v] = 0.f; acc01.f[v] = 0.f; acc10.f[v] = 0.f; acc11.f[v] = 0.f;
    }

#pragma unroll 2
    for (int ch = 0; ch < NCHUNK; ch++) {
        if (ch + 1 < NCHUNK) __builtin_prefetch(wp0 + (ch + 1) * 32, 0, 0); // global_prefetch_b8
        AFrag a0, a1, bf0, bf1;
        bf0.v = wp0[ch * 32];
        bf1.v = wp1[ch * 32];
        int kb = ch * 32;
        if constexpr (CILOG2 >= 5) {
            // CI >= 32: whole chunk shares one tap, k always < K
            int tap = kb >> CILOG2;
            int cib = kb & (CI - 1);
            int kd = tap / 9;
            int r2 = tap - kd * 9;
            int kh = r2 / 3, kw = r2 - kh * 3;
            size_t toff = (size_t)kd * PHW + kh * PW + kw;
#pragma unroll
            for (int j = 0; j < 16; j++) {
                int klocal = ((j >> 3) << 4) + khalf + (j & 7);
                size_t coff = (size_t)(cib + klocal) * DINPHW + toff;
                a0.h[j] = v0 ? in[lanebase0 + coff] : (_Float16)0.f;
                a1.h[j] = v1 ? in[lanebase1 + coff] : (_Float16)0.f;
            }
        } else {
            // CI == 16 (layer 0): taps may change within a chunk, tail may exceed K
#pragma unroll
            for (int j = 0; j < 16; j++) {
                int klocal = ((j >> 3) << 4) + khalf + (j & 7);
                int k = kb + klocal;
                _Float16 x0 = (_Float16)0.f, x1 = (_Float16)0.f;
                if (k < K) {
                    int ci = k & (CI - 1);
                    int tap = k >> CILOG2;
                    int kd = tap / 9;
                    int r2 = tap - kd * 9;
                    int kh = r2 / 3, kw = r2 - kh * 3;
                    size_t coff = (size_t)ci * DINPHW + (size_t)kd * PHW + kh * PW + kw;
                    if (v0) x0 = in[lanebase0 + coff];
                    if (v1) x1 = in[lanebase1 + coff];
                }
                a0.h[j] = x0;
                a1.h[j] = x1;
            }
        }
        acc00.v = __builtin_amdgcn_wmma_f32_16x16x32_f16(false, a0.v, false, bf0.v, (short)0, acc00.v, false, false);
        acc01.v = __builtin_amdgcn_wmma_f32_16x16x32_f16(false, a0.v, false, bf1.v, (short)0, acc01.v, false, false);
        acc10.v = __builtin_amdgcn_wmma_f32_16x16x32_f16(false, a1.v, false, bf0.v, (short)0, acc10.v, false, false);
        acc11.v = __builtin_amdgcn_wmma_f32_16x16x32_f16(false, a1.v, false, bf1.v, (short)0, acc11.v, false, false);
    }

    float bv0 = bias[co0];
    float bv1 = bias[co1];
    size_t ob0 = (((size_t)b * CO + co0) * DOUT + d) * HW;
    size_t ob1 = (((size_t)b * CO + co1) * DOUT + d) * HW;
#pragma unroll
    for (int v = 0; v < 8; v++) {
        int m  = v + khalf;               // D row M = v + 8*(lane/16)
        int p0 = tilex * 32 + m;
        int p1 = p0 + 16;
        if (p0 < HW) {
            out[ob0 + p0] = acc00.f[v] + bv0;
            out[ob1 + p0] = acc01.f[v] + bv1;
        }
        if (p1 < HW) {
            out[ob0 + p1] = acc10.f[v] + bv0;
            out[ob1 + p1] = acc11.f[v] + bv1;
        }
    }
}

// ---------------- per-channel BN stats over (B, D, H, W) ----------------
__global__ void chan_stats_kernel(const float* __restrict__ x, float* __restrict__ mean,
                                  float* __restrict__ istd, int Co, int Dout) {
    int c = blockIdx.x;
    __shared__ float s1[256], s2[256];
    int DHW = Dout * HW;
    float a = 0.f, a2 = 0.f;
    for (int i = threadIdx.x; i < BATCH * DHW; i += blockDim.x) {
        int b = i / DHW, r = i % DHW;
        float v = x[((size_t)b * Co + c) * DHW + r];
        a += v; a2 += v * v;
    }
    s1[threadIdx.x] = a; s2[threadIdx.x] = a2;
    __syncthreads();
    for (int off = 128; off > 0; off >>= 1) {
        if ((int)threadIdx.x < off) {
            s1[threadIdx.x] += s1[threadIdx.x + off];
            s2[threadIdx.x] += s2[threadIdx.x + off];
        }
        __syncthreads();
    }
    if (threadIdx.x == 0) {
        float cnt = (float)(BATCH * DHW);
        float m = s1[0] / cnt;
        float var = s2[0] / cnt - m * m;
        if (var < 0.f) var = 0.f;
        mean[c] = m;
        istd[c] = rsqrtf(var + EPSV);
    }
}

// ---------------- BN + relu, write padded f16 for next conv ----------------
__global__ void bnrelu_f16_kernel(const float* __restrict__ x, const float* __restrict__ mean,
                                  const float* __restrict__ istd, const float* __restrict__ g,
                                  const float* __restrict__ be, _Float16* __restrict__ y,
                                  int Co, int Dout) {
    int i = blockIdx.x * blockDim.x + threadIdx.x;
    int DHW = Dout * HW;
    int total = BATCH * Co * DHW;
    if (i >= total) return;
    int c = (i / DHW) % Co;
    int b = i / (Co * DHW);
    int r = i % DHW;
    int dd = r / HW;
    int rest = r % HW;
    int ie = rest / AZI_N, ia = rest % AZI_N;
    float v = (x[i] - mean[c]) * istd[c] * g[c] + be[c];
    _Float16 hv = (_Float16)fmaxf(v, 0.f);
    size_t base = (((size_t)b * Co + c) * Dout + dd) * PHW;
    y[base + (size_t)(ie + 1) * PW + (ia + 1)] = hv;
    if (ia == 0)           y[base + (size_t)(ie + 1) * PW + (PW - 1)] = hv;
    if (ia == AZI_N - 1)   y[base + (size_t)(ie + 1) * PW + 0] = hv;
}

// ---------------- final: BN + relu + max over (H,W) ----------------
__global__ void final_max_kernel(const float* __restrict__ x, const float* __restrict__ mean,
                                 const float* __restrict__ istd, const float* __restrict__ g,
                                 const float* __restrict__ be, float* __restrict__ out) {
    int blk = blockIdx.x;                 // b*32 + c
    int b = blk >> 5, c = blk & 31;
    __shared__ float sm[256];
    const float* xc = x + ((size_t)b * 32 + c) * HW;
    float m = mean[c], sc = istd[c] * g[c], bec = be[c];
    float mx = 0.f;
    for (int i = threadIdx.x; i < HW; i += blockDim.x) {
        float v = fmaxf((xc[i] - m) * sc + bec, 0.f);
        mx = fmaxf(mx, v);
    }
    sm[threadIdx.x] = mx;
    __syncthreads();
    for (int off = 128; off > 0; off >>= 1) {
        if ((int)threadIdx.x < off)
            sm[threadIdx.x] = fmaxf(sm[threadIdx.x], sm[threadIdx.x + off]);
        __syncthreads();
    }
    if (threadIdx.x == 0) out[blk] = sm[0];
}

// ---------------- host launcher ----------------
extern "C" void kernel_launch(void* const* d_in, const int* in_sizes, int n_in,
                              void* d_out, int out_size, void* d_ws, size_t ws_size,
                              hipStream_t stream) {
    (void)in_sizes; (void)n_in; (void)out_size; (void)ws_size;

    const float* x   = (const float*)d_in[0];
    const float* wr  = (const float*)d_in[1];
    const float* br  = (const float*)d_in[2];
    const float* gr  = (const float*)d_in[3];
    const float* ber = (const float*)d_in[4];

    char* ws = (char*)d_ws;
    float*    Rm      = (float*)(ws + OFF_R);
    float*    psum    = (float*)(ws + OFF_PSUM);
    float*    rss     = (float*)(ws + OFF_RSS);
    float*    meanb   = (float*)(ws + OFF_MEAN);
    float*    istdb   = (float*)(ws + OFF_ISTD);
    float*    drot    = (float*)(ws + OFF_DROT);
    float*    newp    = (float*)(ws + OFF_NEWP);
    _Float16* act0    = (_Float16*)(ws + OFF_ACT0);
    _Float16* act1    = (_Float16*)(ws + OFF_ACT1);
    float*    convout = (float*)(ws + OFF_CONV);
    _Float16* wpk     = (_Float16*)(ws + OFF_WPK);

    zero_kernel<<<1, 32, 0, stream>>>(psum, 9);
    cov_eig_kernel<<<BATCH, 256, 0, stream>>>(x, Rm);
    rotate_kernel<<<(BATCH * NPTS + 255) / 256, 256, 0, stream>>>(x, Rm, drot);
    group_kernel<<<BATCH * S_VOX, 32, 0, stream>>>(drot, newp, psum);
    raise_stats_kernel<<<1, 16, 0, stream>>>(psum, wr, br, gr, ber, rss);
    raise_max_kernel<<<(BATCH * S_VOX * 16 + 255) / 256, 256, 0, stream>>>(newp, wr, br, rss, act0);

    struct Layer { int wi, Ci, ciLog2, Co, Din, Dout, KD; };
    const Layer L[6] = {
        {5,  16,  4,  32, 9, 7, 3},
        {9,  32,  5,  64, 7, 5, 3},
        {13, 64,  6,  64, 5, 3, 3},
        {17, 64,  6, 128, 3, 1, 3},
        {21, 128, 7, 128, 1, 1, 1},   // 2D conv (KD=1)
        {25, 128, 7,  32, 1, 1, 1},   // 2D conv (KD=1)
    };

    _Float16* cur = act0;
    _Float16* nxt = act1;
    for (int li = 0; li < 6; li++) {
        const float* wl  = (const float*)d_in[L[li].wi + 0];
        const float* bl  = (const float*)d_in[L[li].wi + 1];
        const float* gl  = (const float*)d_in[L[li].wi + 2];
        const float* bel = (const float*)d_in[L[li].wi + 3];
        int K9 = L[li].KD * 9;
        int K = L[li].Ci * K9;
        int nchunk = (K + 31) / 32;
        int wtotal = (L[li].Co / 16) * nchunk * 512;

        wpack_kernel<<<(wtotal + 255) / 256, 256, 0, stream>>>(
            wl, wpk, L[li].Ci, L[li].ciLog2, K9, nchunk, wtotal);

        int padCD = L[li].Ci * L[li].Din;
        padzero_kernel<<<(BATCH * padCD * 2 * PW + 255) / 256, 256, 0, stream>>>(cur, padCD);

        dim3 grid(57, L[li].Co / 32, BATCH * L[li].Dout);
        const v16h* wv = (const v16h*)wpk;
        switch (li) {
            case 0: conv_wmma_kernel<16, 4,  32, 9, 7, 3><<<grid, 32, 0, stream>>>(cur, wv, bl, convout); break;
            case 1: conv_wmma_kernel<32, 5,  64, 7, 5, 3><<<grid, 32, 0, stream>>>(cur, wv, bl, convout); break;
            case 2: conv_wmma_kernel<64, 6,  64, 5, 3, 3><<<grid, 32, 0, stream>>>(cur, wv, bl, convout); break;
            case 3: conv_wmma_kernel<64, 6, 128, 3, 1, 3><<<grid, 32, 0, stream>>>(cur, wv, bl, convout); break;
            case 4: conv_wmma_kernel<128, 7, 128, 1, 1, 1><<<grid, 32, 0, stream>>>(cur, wv, bl, convout); break;
            case 5: conv_wmma_kernel<128, 7,  32, 1, 1, 1><<<grid, 32, 0, stream>>>(cur, wv, bl, convout); break;
        }

        chan_stats_kernel<<<L[li].Co, 256, 0, stream>>>(convout, meanb, istdb,
                                                        L[li].Co, L[li].Dout);
        if (li < 5) {
            int total = BATCH * L[li].Co * L[li].Dout * HW;
            bnrelu_f16_kernel<<<(total + 255) / 256, 256, 0, stream>>>(
                convout, meanb, istdb, gl, bel, nxt, L[li].Co, L[li].Dout);
            _Float16* t = cur; cur = nxt; nxt = t;
        } else {
            final_max_kernel<<<BATCH * 32, 256, 0, stream>>>(
                convout, meanb, istdb, gl, bel, (float*)d_out);
        }
    }
}
